// DiscriminatorStack_11759620457092
// MI455X (gfx1250) — matrix operand order repaired and checked by reading, not compiled, not run on hardware
//
#include <hip/hip_runtime.h>
#include <hip/hip_bf16.h>

// ---------------------------------------------------------------------------
// Types for CDNA5 WMMA (wave32, v_wmma_f32_16x16x32_bf16)
// ---------------------------------------------------------------------------
typedef __attribute__((ext_vector_type(16))) __bf16 v16bf;
typedef __attribute__((ext_vector_type(8)))  float  v8f;

__device__ __forceinline__ unsigned short f2bf(float f) {
    union { float f; unsigned int u; } x; x.f = f;
    unsigned int u = x.u;
    unsigned int r = u + 0x7FFFu + ((u >> 16) & 1u);   // round-to-nearest-even
    return (unsigned short)(r >> 16);
}
__device__ __forceinline__ __bf16 bfbits(unsigned short s) {
    union { unsigned short s; __bf16 b; } x; x.s = s; return x.b;
}
__device__ __forceinline__ float lrelu(float v) { return v >= 0.f ? v : 0.2f * v; }

// ---------------------------------------------------------------------------
// Elementwise conversion kernels
// ---------------------------------------------------------------------------
__global__ __launch_bounds__(256) void cvt_bf16(const float* __restrict__ in,
                                                unsigned short* __restrict__ out,
                                                size_t n) {
    size_t i = (size_t)blockIdx.x * blockDim.x + threadIdx.x;
    if (i < n) out[i] = f2bf(in[i]);
}

// T[k][j] (k<Kd, j<Nn) -> Tt[j][k]  (so MBD GEMM is NT, K-contiguous)
__global__ __launch_bounds__(256) void cvt_T(const float* __restrict__ T,
                                             unsigned short* __restrict__ Tt,
                                             int Kd, int Nn) {
    size_t i = (size_t)blockIdx.x * blockDim.x + threadIdx.x;
    if (i >= (size_t)Kd * Nn) return;
    int k = (int)(i / Nn);
    int j = (int)(i - (size_t)k * Nn);
    Tt[(size_t)j * Kd + k] = f2bf(T[i]);
}

// image [128,3,64,64] f32 -> xcat channels 0..2 (bf16, NCHW with C=4)
__global__ __launch_bounds__(256) void pack_image(const float* __restrict__ img,
                                                  unsigned short* __restrict__ xcat) {
    size_t i = (size_t)blockIdx.x * blockDim.x + threadIdx.x;
    const size_t total = (size_t)128 * 3 * 4096;
    if (i >= total) return;
    int b = (int)(i / (3 * 4096));
    int r = (int)(i - (size_t)b * (3 * 4096));
    int c = r / 4096;
    int p = r - c * 4096;
    xcat[((size_t)b * 4 + c) * 4096 + p] = f2bf(img[i]);
}

// ---------------------------------------------------------------------------
// Generic NT bf16 WMMA GEMM:  C[M,N] = A[M,K] * B[N,K]^T  (f32 accumulate)
//   BMODE 0: B is plain [N,K] row-major bf16
//   BMODE 1: B is implicit im2col of NCHW bf16 tensor (4x4 kernel, stride 2,
//            pad 1); n=(b,oh,ow), k=(ic,kh,kw) with k = ic*16 + kh*4 + kw
//   EMODE 0: plain f32 store C[m*N+n]               (BN staging / Ms)
//   EMODE 1: ReLU, bf16 -> xcat channel 3           (FC; m=batch, n=pixel)
//   EMODE 2: LReLU, bf16 -> NCHW                    (conv1)
//   EMODE 4: conv5: f32 feat -> d_out[:, :1024] and bf16 copy -> featbf
// Tile: 128(M) x 64(N) x 64(K), 256 threads = 8 waves.
// Per K-stage each wave issues 8 WMMAs (A-fragment reused across 4 N-subtiles
// per 32-wide k-half), so barrier/staging cost is amortized 2x vs TK=32.
// ---------------------------------------------------------------------------
template <int BMODE, int EMODE>
__global__ __launch_bounds__(256) void gemm_bf16_wmma(
    const unsigned short* __restrict__ A,
    const unsigned short* __restrict__ B,
    int M, int N, int K,
    int IC, int IH, int IW, int OH, int OW,
    float* __restrict__ c32, unsigned short* __restrict__ cbf,
    float* __restrict__ dout, unsigned short* __restrict__ featbf) {

    constexpr int TM = 128, TN = 64, TK = 64;
    constexpr int LDA = TK + 2;   // padded ushort rows
    constexpr int LDB = TK + 2;
    __shared__ unsigned short lsA[TM * LDA];
    __shared__ unsigned short lsB[TN * LDB];

    const int tid  = threadIdx.x;
    const int wave = tid >> 5;
    const int lane = tid & 31;
    const int h    = lane >> 4;     // lane-half
    const int ml   = lane & 15;
    const int m0   = blockIdx.y * TM;
    const int n0   = blockIdx.x * TN;
    const int mbase = wave * 16;
    const int HWo  = OH * OW;

    v8f acc[4];
#pragma unroll
    for (int i = 0; i < 4; i++)
#pragma unroll
        for (int j = 0; j < 8; j++) acc[i][j] = 0.0f;

    // staging roles (fixed per thread)
    const int arow = tid >> 1;              // 128 rows, 2 threads/row
    const int acb  = (tid & 1) << 5;        // 32 contiguous k-elems each
    const int brow = tid >> 2;              // 64 n-rows, 4 threads/row
    const int bcb  = (tid & 3) << 4;        // 16 k-elems each

    for (int k0 = 0; k0 < K; k0 += TK) {
        // ---- prefetch next K-tile (gfx1250 global_prefetch_b8) ----
        if (k0 + TK < K) {
            __builtin_prefetch(A + (size_t)(m0 + arow) * K + k0 + TK + acb, 0, 1);
            if (BMODE == 0)
                __builtin_prefetch(B + (size_t)(n0 + brow) * K + k0 + TK + bcb, 0, 1);
        }
        // ---- stage A tile (TM x TK) ----
        {
            const unsigned short* src = A + (size_t)(m0 + arow) * K + k0 + acb;
            unsigned short* dst = lsA + arow * LDA + acb;
#pragma unroll
            for (int i = 0; i < 32; i++) dst[i] = src[i];
        }
        // ---- stage B tile (TN x TK), n-major / k-contiguous ----
        {
            unsigned short* dst = lsB + brow * LDB + bcb;
            if (BMODE == 0) {
                const unsigned short* src = B + (size_t)(n0 + brow) * K + k0 + bcb;
#pragma unroll
                for (int i = 0; i < 16; i++) dst[i] = src[i];
            } else {
                int n   = n0 + brow;
                int b   = n / HWo;
                int pix = n - b * HWo;
                int oh  = pix / OW;
                int ow  = pix - oh * OW;
#pragma unroll
                for (int i = 0; i < 16; i++) {
                    int k  = k0 + bcb + i;
                    int ic = k >> 4;
                    int r  = k & 15;
                    int kh = r >> 2;
                    int kw = r & 3;
                    int ih = oh * 2 - 1 + kh;
                    int iw = ow * 2 - 1 + kw;
                    unsigned short v = 0;
                    if (ih >= 0 && ih < IH && iw >= 0 && iw < IW)
                        v = B[(size_t)((b * IC + ic) * IH + ih) * IW + iw];
                    dst[i] = v;
                }
            }
        }
        __syncthreads();

#pragma unroll
        for (int kk = 0; kk < TK; kk += 32) {
            // ---- A fragment (16x32 MxK, ISA per-lane layout) ----
            v16bf afr;
            {
                const unsigned short* ar = lsA + (mbase + ml) * LDA + kk;
#pragma unroll
                for (int v = 0; v < 8; v++) {
                    int kb = ((v < 4) ? (2 * v) : (16 + 2 * (v - 4))) + 8 * h;
                    afr[2 * v]     = bfbits(ar[kb]);
                    afr[2 * v + 1] = bfbits(ar[kb + 1]);
                }
            }
            // ---- 4 N-subtiles: B fragment in same k order as A (8h+e, 16+8h+e-8) ----
#pragma unroll
            for (int ns = 0; ns < 4; ns++) {
                const unsigned short* br = lsB + (ns * 16 + ml) * LDB + kk + 8 * h;
                v16bf bfr;
#pragma unroll
                for (int e = 0; e < 16; e++) bfr[e] = bfbits(br[(e < 8) ? e : (e + 8)]);
                acc[ns] = __builtin_amdgcn_wmma_f32_16x16x32_bf16(
                    false, afr, false, bfr, (short)0, acc[ns], false, false);
            }
        }
        __syncthreads();
    }

    // ---- epilogue: C/D layout: VGPR r -> M = r + 8*h, N = lane&15 ----
#pragma unroll
    for (int ns = 0; ns < 4; ns++) {
#pragma unroll
        for (int r = 0; r < 8; r++) {
            float v = acc[ns][r];
            int m = m0 + mbase + r + 8 * h;
            int n = n0 + ns * 16 + ml;
            if (EMODE == 0) {
                c32[(size_t)m * N + n] = v;
            } else if (EMODE == 1) {               // FC: relu -> xcat ch3
                v = v > 0.f ? v : 0.f;
                cbf[((size_t)m * 4 + 3) * 4096 + n] = f2bf(v);
            } else if (EMODE == 2) {               // conv1: lrelu -> NCHW bf16
                v = lrelu(v);
                int b = n / HWo; int pix = n - b * HWo;
                cbf[((size_t)(b * M + m)) * HWo + pix] = f2bf(v);
            } else {                               // conv5: feat -> d_out + bf16
                int b = n >> 2; int pos = n & 3;
                dout[(size_t)b * 1280 + m * 4 + pos] = v;
                featbf[(size_t)b * 1024 + m * 4 + pos] = f2bf(v);
            }
        }
    }
}

// ---------------------------------------------------------------------------
// BatchNorm (training stats over N,H,W per channel) on CNHW f32 staging
// ---------------------------------------------------------------------------
__global__ __launch_bounds__(256) void bn_stats(const float* __restrict__ stage,
                                                int Ndim,
                                                float* __restrict__ mean,
                                                float* __restrict__ rstd) {
    int oc = blockIdx.x;
    const float* row = stage + (size_t)oc * Ndim;
    float s = 0.f, s2 = 0.f;
    for (int i = threadIdx.x; i < Ndim; i += 256) {
        float v = row[i]; s += v; s2 += v * v;
    }
    __shared__ float sh[256], sh2[256];
    sh[threadIdx.x] = s; sh2[threadIdx.x] = s2;
    __syncthreads();
    for (int off = 128; off > 0; off >>= 1) {
        if (threadIdx.x < off) {
            sh[threadIdx.x]  += sh[threadIdx.x + off];
            sh2[threadIdx.x] += sh2[threadIdx.x + off];
        }
        __syncthreads();
    }
    if (threadIdx.x == 0) {
        float mn  = sh[0] / (float)Ndim;
        float var = sh2[0] / (float)Ndim - mn * mn;
        mean[oc] = mn;
        rstd[oc] = rsqrtf(var + 1e-5f);
    }
}

__global__ __launch_bounds__(256) void bn_apply(const float* __restrict__ stage,
                                                const float* __restrict__ mean,
                                                const float* __restrict__ rstd,
                                                const float* __restrict__ gamma,
                                                const float* __restrict__ beta,
                                                unsigned short* __restrict__ out,
                                                int OC, int HW, int Ndim) {
    size_t idx = (size_t)blockIdx.x * blockDim.x + threadIdx.x;
    size_t total = (size_t)OC * Ndim;
    if (idx >= total) return;
    int oc = (int)(idx / Ndim);
    int n  = (int)(idx - (size_t)oc * Ndim);
    float v = stage[idx];
    v = (v - mean[oc]) * rstd[oc] * gamma[oc] + beta[oc];
    v = lrelu(v);
    int b = n / HW; int pix = n - b * HW;
    out[((size_t)(b * OC + oc)) * HW + pix] = f2bf(v);
}

// ---------------------------------------------------------------------------
// Minibatch discrimination: mbd[i, t] = sum_j exp(-||Ms[i,t,:]-Ms[j,t,:]||_1)
// Ms: [128, 256, 16] f32. One block per i, thread t = output channel.
// ---------------------------------------------------------------------------
__global__ __launch_bounds__(256) void mbd_kernel(const float* __restrict__ Ms,
                                                  float* __restrict__ dout) {
    __shared__ float sh[4096];
    int i = blockIdx.x;
    int t = threadIdx.x;
    float mi[16];
#pragma unroll
    for (int c = 0; c < 16; c++) mi[c] = Ms[(size_t)i * 4096 + t * 16 + c];
    float acc = 0.f;
    for (int j = 0; j < 128; j++) {
        for (int e = t; e < 4096; e += 256) sh[e] = Ms[(size_t)j * 4096 + e];
        __syncthreads();
        float d = 0.f;
#pragma unroll
        for (int c = 0; c < 16; c++) d += fabsf(mi[c] - sh[t * 16 + c]);
        acc += __expf(-d);
        __syncthreads();
    }
    dout[(size_t)i * 1280 + 1024 + t] = acc;
}

// ---------------------------------------------------------------------------
// Host-side orchestration
// ---------------------------------------------------------------------------
extern "C" void kernel_launch(void* const* d_in, const int* in_sizes, int n_in,
                              void* d_out, int out_size, void* d_ws, size_t ws_size,
                              hipStream_t stream) {
    const float* image = (const float*)d_in[0];
    const float* wordv = (const float*)d_in[1];
    const float* fcw   = (const float*)d_in[2];
    const float* w1    = (const float*)d_in[3];
    const float* w2    = (const float*)d_in[4];
    const float* w3    = (const float*)d_in[5];
    const float* w4    = (const float*)d_in[6];
    const float* w5    = (const float*)d_in[7];
    const float* g2    = (const float*)d_in[8];
    const float* b2    = (const float*)d_in[9];
    const float* g3    = (const float*)d_in[10];
    const float* b3    = (const float*)d_in[11];
    const float* g4    = (const float*)d_in[12];
    const float* b4    = (const float*)d_in[13];
    const float* T     = (const float*)d_in[14];
    float* out = (float*)d_out;

    char* ws = (char*)d_ws;
    size_t off = 0;
    auto alloc = [&](size_t bytes) -> char* {
        char* p = ws + off;
        off += (bytes + 255) & ~(size_t)255;
        return p;
    };

    // bf16 weights / activations in workspace
    unsigned short* fcw_bf = (unsigned short*)alloc((size_t)16777216 * 2);
    unsigned short* wv_bf  = (unsigned short*)alloc((size_t)524288 * 2);
    unsigned short* w1_bf  = (unsigned short*)alloc((size_t)8192 * 2);
    unsigned short* w2_bf  = (unsigned short*)alloc((size_t)524288 * 2);
    unsigned short* w3_bf  = (unsigned short*)alloc((size_t)1048576 * 2);
    unsigned short* w4_bf  = (unsigned short*)alloc((size_t)2097152 * 2);
    unsigned short* w5_bf  = (unsigned short*)alloc((size_t)2097152 * 2);
    unsigned short* Tt_bf  = (unsigned short*)alloc((size_t)4194304 * 2);
    unsigned short* xcat   = (unsigned short*)alloc((size_t)2097152 * 2);  // [128,4,64,64]
    unsigned short* x1     = (unsigned short*)alloc((size_t)16777216 * 2); // [128,128,32,32]
    unsigned short* x2     = (unsigned short*)alloc((size_t)8388608 * 2);  // [128,256,16,16]
    unsigned short* x3     = (unsigned short*)alloc((size_t)2097152 * 2);  // [128,256,8,8]
    unsigned short* x4     = (unsigned short*)alloc((size_t)1048576 * 2);  // [128,512,4,4]
    unsigned short* featbf = (unsigned short*)alloc((size_t)131072 * 2);   // [128,1024]
    float* stage = (float*)alloc((size_t)8388608 * 4);  // max CNHW: 256 x 32768
    float* Ms    = (float*)alloc((size_t)524288 * 4);   // [128,4096]
    float* meanb = (float*)alloc((size_t)512 * 4);
    float* rstdb = (float*)alloc((size_t)512 * 4);

    auto nb = [](size_t n) { return (unsigned)((n + 255) / 256); };

    // ---- f32 -> bf16 conversions ----
    cvt_bf16<<<nb(16777216), 256, 0, stream>>>(fcw, fcw_bf, 16777216);
    cvt_bf16<<<nb(524288),  256, 0, stream>>>(wordv, wv_bf, 524288);
    cvt_bf16<<<nb(8192),    256, 0, stream>>>(w1, w1_bf, 8192);
    cvt_bf16<<<nb(524288),  256, 0, stream>>>(w2, w2_bf, 524288);
    cvt_bf16<<<nb(1048576), 256, 0, stream>>>(w3, w3_bf, 1048576);
    cvt_bf16<<<nb(2097152), 256, 0, stream>>>(w4, w4_bf, 2097152);
    cvt_bf16<<<nb(2097152), 256, 0, stream>>>(w5, w5_bf, 2097152);
    cvt_T<<<nb(4194304), 256, 0, stream>>>(T, Tt_bf, 1024, 4096);
    pack_image<<<nb(1572864), 256, 0, stream>>>(image, xcat);

    // ---- FC: [128,4096] = wv * fcw^T, ReLU -> xcat channel 3 ----
    gemm_bf16_wmma<0, 1><<<dim3(4096 / 64, 1), 256, 0, stream>>>(
        wv_bf, fcw_bf, 128, 4096, 4096,
        0, 0, 0, 1, 1, nullptr, xcat, nullptr, nullptr);

    // ---- conv1: 4->128, 64->32, LReLU -> x1 ----
    gemm_bf16_wmma<1, 2><<<dim3(131072 / 64, 1), 256, 0, stream>>>(
        w1_bf, xcat, 128, 131072, 64,
        4, 64, 64, 32, 32, nullptr, x1, nullptr, nullptr);

    // ---- conv2: 128->256, 32->16, BN+LReLU -> x2 ----
    gemm_bf16_wmma<1, 0><<<dim3(32768 / 64, 2), 256, 0, stream>>>(
        w2_bf, x1, 256, 32768, 2048,
        128, 32, 32, 16, 16, stage, nullptr, nullptr, nullptr);
    bn_stats<<<256, 256, 0, stream>>>(stage, 32768, meanb, rstdb);
    bn_apply<<<nb((size_t)256 * 32768), 256, 0, stream>>>(
        stage, meanb, rstdb, g2, b2, x2, 256, 256, 32768);

    // ---- conv3: 256->256, 16->8, BN+LReLU -> x3 ----
    gemm_bf16_wmma<1, 0><<<dim3(8192 / 64, 2), 256, 0, stream>>>(
        w3_bf, x2, 256, 8192, 4096,
        256, 16, 16, 8, 8, stage, nullptr, nullptr, nullptr);
    bn_stats<<<256, 256, 0, stream>>>(stage, 8192, meanb, rstdb);
    bn_apply<<<nb((size_t)256 * 8192), 256, 0, stream>>>(
        stage, meanb, rstdb, g3, b3, x3, 256, 64, 8192);

    // ---- conv4: 256->512, 8->4, BN+LReLU -> x4 ----
    gemm_bf16_wmma<1, 0><<<dim3(2048 / 64, 4), 256, 0, stream>>>(
        w4_bf, x3, 512, 2048, 4096,
        256, 8, 8, 4, 4, stage, nullptr, nullptr, nullptr);
    bn_stats<<<512, 256, 0, stream>>>(stage, 2048, meanb, rstdb);
    bn_apply<<<nb((size_t)512 * 2048), 256, 0, stream>>>(
        stage, meanb, rstdb, g4, b4, x4, 512, 16, 2048);

    // ---- conv5: 512->256, 4->2, feat -> d_out[:, :1024] + featbf ----
    gemm_bf16_wmma<1, 4><<<dim3(512 / 64, 2), 256, 0, stream>>>(
        w5_bf, x4, 256, 512, 8192,
        512, 4, 4, 2, 2, nullptr, nullptr, out, featbf);

    // ---- MBD matmul: Ms[128,4096] = feat * T ----
    gemm_bf16_wmma<0, 0><<<dim3(4096 / 64, 1), 256, 0, stream>>>(
        featbf, Tt_bf, 128, 4096, 1024,
        0, 0, 0, 1, 1, Ms, nullptr, nullptr, nullptr);

    // ---- pairwise L1 / exp / batch-sum -> d_out[:, 1024:1280] ----
    mbd_kernel<<<128, 256, 0, stream>>>(Ms, out);
}
